// D_KGTransformer_6554120094294
// MI455X (gfx1250) — matrix operand order repaired and checked by reading, not compiled, not run on hardware
//
#include <hip/hip_runtime.h>

typedef __attribute__((ext_vector_type(4)))  _Float16 v4h;
typedef __attribute__((ext_vector_type(8)))  _Float16 v8h;
typedef __attribute__((ext_vector_type(16))) _Float16 v16h;
typedef __attribute__((ext_vector_type(8)))  float    v8f;

union U16h { v16h v; v8h h[2]; };

// ---------------- utility kernels ----------------

__global__ __launch_bounds__(256) void k_zero_f32(float4* __restrict__ p, long long n4) {
    long long i = (long long)blockIdx.x * blockDim.x + threadIdx.x;
    if (i < n4) p[i] = make_float4(0.f, 0.f, 0.f, 0.f);
}

// scatter feat[:, :2, :] rows into A0[node, :] with atomic f32 adds
__global__ __launch_bounds__(256) void k_scatter_feat(
    const float* __restrict__ feat, const long long* __restrict__ nid,
    float* __restrict__ A0, int T, int d) {
    long long i = (long long)blockIdx.x * blockDim.x + threadIdx.x;
    long long total = (long long)2 * T * d;
    if (i >= total) return;
    int c = (int)(i % d);
    long long p = i / d;
    long long t = p >> 1;
    int j = (int)(p & 1);
    long long row = nid[t * 7 + j];
    float v = feat[((size_t)t * 7 + j) * d + c];
    atomicAdd(&A0[(size_t)row * d + c], v);
}

// densify COO adjacency into L [N x N]
__global__ __launch_bounds__(256) void k_build_L(
    const int* __restrict__ rows, const int* __restrict__ cols,
    const float* __restrict__ vals, float* __restrict__ L, int E, int N) {
    int e = blockIdx.x * blockDim.x + threadIdx.x;
    if (e < E) atomicAdd(&L[(size_t)rows[e] * N + cols[e]], vals[e]);
}

// straight f32 -> f16 convert (vectorized: float4 in, 4x f16 out)
__global__ __launch_bounds__(256) void k_cvt_f16(
    const float* __restrict__ src, _Float16* __restrict__ dst, long long n4) {
    long long i = (long long)blockIdx.x * blockDim.x + threadIdx.x;
    if (i >= n4) return;
    float4 v = ((const float4*)src)[i];
    v4h h = { (_Float16)v.x, (_Float16)v.y, (_Float16)v.z, (_Float16)v.w };
    ((v4h*)dst)[i] = h;
}

// B [K x N] f32  ->  BT [N x K] f16  (LDS-tiled transpose, 32x32 tiles)
__global__ __launch_bounds__(256) void k_cvtT_f16(
    const float* __restrict__ B, _Float16* __restrict__ BT, int K, int N) {
    __shared__ _Float16 t[32][33];
    int tx = threadIdx.x & 31;
    int ty = threadIdx.x >> 5; // 0..7
    int k0 = blockIdx.y * 32, n0 = blockIdx.x * 32;
#pragma unroll
    for (int i = 0; i < 32; i += 8)
        t[ty + i][tx] = (_Float16)B[(size_t)(k0 + ty + i) * N + (n0 + tx)];
    __syncthreads();
#pragma unroll
    for (int i = 0; i < 32; i += 8)
        BT[(size_t)(n0 + ty + i) * K + (k0 + tx)] = t[tx][ty + i];
}

// ---------------- WMMA GEMM ----------------
// C[M x N] f32 = A[M x K] f16 * BT[N x K]^T f16 (+ bias[n]) (+ relu)
// block = 128 threads (4 waves, 2x2), block tile 64x64, wave tile 32x32 (2x2 of 16x16)
__global__ __launch_bounds__(128) void k_gemm_f16w(
    const _Float16* __restrict__ A, const _Float16* __restrict__ BT,
    const float* __restrict__ bias, float* __restrict__ C,
    int M, int K, int N, int relu) {
    const int lane = threadIdx.x & 31;
    const int wave = threadIdx.x >> 5;
    const int wm = wave >> 1, wn = wave & 1;
    const int tM = blockIdx.y * 64 + wm * 32;
    const int tN = blockIdx.x * 64 + wn * 32;
    const int l16 = lane & 15;
    const int half = lane >> 4;

    v8f acc[2][2];
#pragma unroll
    for (int j = 0; j < 2; ++j) {
        float bv = bias ? bias[tN + j * 16 + l16] : 0.0f;
        v8f c;
#pragma unroll
        for (int r = 0; r < 8; ++r) c[r] = bv;
        acc[0][j] = c;
        acc[1][j] = c;
    }

    // A fragment (16-bit 16x32 layout): lane half h covers K = h*16..h*16+15, contiguous (matches B)
    const _Float16* a0p = A + (size_t)(tM + l16) * K;
    const _Float16* a1p = A + (size_t)(tM + 16 + l16) * K;
    // B fragment (32x16 layout): lane half h covers K = h*16..h*16+15, contiguous in BT row
    const _Float16* b0p = BT + (size_t)(tN + l16) * K;
    const _Float16* b1p = BT + (size_t)(tN + 16 + l16) * K;
    const int aoff = half * 16;
    const int boff = half * 16;

#pragma unroll 2
    for (int k = 0; k < K; k += 32) {
        U16h a0, a1;
        a0.h[0] = *(const v8h*)(a0p + k + aoff);
        a0.h[1] = *(const v8h*)(a0p + k + 8 + aoff);
        a1.h[0] = *(const v8h*)(a1p + k + aoff);
        a1.h[1] = *(const v8h*)(a1p + k + 8 + aoff);
        v16h b0 = *(const v16h*)(b0p + k + boff);
        v16h b1 = *(const v16h*)(b1p + k + boff);
        acc[0][0] = __builtin_amdgcn_wmma_f32_16x16x32_f16(false, a0.v, false, b0, (short)0, acc[0][0], false, false);
        acc[0][1] = __builtin_amdgcn_wmma_f32_16x16x32_f16(false, a0.v, false, b1, (short)0, acc[0][1], false, false);
        acc[1][0] = __builtin_amdgcn_wmma_f32_16x16x32_f16(false, a1.v, false, b0, (short)0, acc[1][0], false, false);
        acc[1][1] = __builtin_amdgcn_wmma_f32_16x16x32_f16(false, a1.v, false, b1, (short)0, acc[1][1], false, false);
    }

    // C/D layout: VGPR r -> (M = r + 8*half, N = l16)
#pragma unroll
    for (int i = 0; i < 2; ++i)
#pragma unroll
        for (int j = 0; j < 2; ++j) {
            int n = tN + j * 16 + l16;
            float* cp = C + (size_t)(tM + i * 16 + half * 8) * N + n;
#pragma unroll
            for (int r = 0; r < 8; ++r) {
                float v = acc[i][j][r];
                if (relu) v = fmaxf(v, 0.0f);
                cp[(size_t)r * N] = v;
            }
        }
}

// ---------------- final gather + swap into output ----------------

__global__ __launch_bounds__(256) void k_assemble(
    const float* __restrict__ feat, const long long* __restrict__ nid,
    const float* __restrict__ LOG, float* __restrict__ out, int T, int d) {
    long long i = (long long)blockIdx.x * blockDim.x + threadIdx.x;
    int d4 = d >> 2;
    long long tot = (long long)T * 7 * d4;
    if (i >= tot) return;
    int c4 = (int)(i % d4);
    long long r = i / d4;
    int s = (int)(r % 7);
    long long t = r / 7;
    float4 v;
    if (s == 0)      v = ((const float4*)(LOG + (size_t)nid[t * 7 + 1] * d))[c4];
    else if (s == 1) v = ((const float4*)(LOG + (size_t)nid[t * 7 + 0] * d))[c4];
    else             v = ((const float4*)(feat + ((size_t)t * 7 + s) * d))[c4];
    ((float4*)out)[i] = v;
}

// ---------------- host launcher ----------------

extern "C" void kernel_launch(void* const* d_in, const int* in_sizes, int n_in,
                              void* d_out, int out_size, void* d_ws, size_t ws_size,
                              hipStream_t stream) {
    const int*       rows = (const int*)d_in[0];
    const int*       cols = (const int*)d_in[1];
    const float*     vals = (const float*)d_in[2];
    const float*     feat = (const float*)d_in[3];
    const long long* nid  = (const long long*)d_in[4];
    const float* W1 = (const float*)d_in[5];
    const float* b1 = (const float*)d_in[6];
    const float* W2 = (const float*)d_in[7];
    const float* b2 = (const float*)d_in[8];
    const float* W3 = (const float*)d_in[9];
    const float* b3 = (const float*)d_in[10];
    float* out = (float*)d_out;

    const int E = in_sizes[0];
    const int N = in_sizes[6];            // 2048
    const int d = in_sizes[10];           // 512
    const int T = in_sizes[3] / (7 * d);  // 32768

    const size_t Nd = (size_t)N * d, NN = (size_t)N * N;
    float* A0f = (float*)d_ws;            // [N x d] scatter buffer, later reused for logits
    float* Lf  = A0f + Nd;                // [N x N] dense adjacency
    float* Xf  = Lf + NN;                 // [N x N] GEMM output ping
    float* Hf  = Xf + NN;                 // [N x N] GEMM output pong
    _Float16* A0h = (_Float16*)(Hf + NN); // f16 mirrors
    _Float16* Lh  = A0h + Nd;
    _Float16* Hh  = Lh + NN;
    _Float16* BT  = Hh + NN;              // [N x K] transposed-B staging
    float* LOG = A0f;

    // zero A0f and Lf (contiguous)
    {
        long long n4 = (long long)(Nd + NN) / 4;
        k_zero_f32<<<(unsigned)((n4 + 255) / 256), 256, 0, stream>>>((float4*)A0f, n4);
    }
    // scatter entity embeddings
    {
        long long tot = (long long)2 * T * d;
        k_scatter_feat<<<(unsigned)((tot + 255) / 256), 256, 0, stream>>>(feat, nid, A0f, T, d);
    }
    // densify adjacency
    k_build_L<<<(E + 255) / 256, 256, 0, stream>>>(rows, cols, vals, Lf, E, N);

    auto cvt = [&](const float* src, _Float16* dst, size_t n) {
        long long n4 = (long long)n / 4;
        k_cvt_f16<<<(unsigned)((n4 + 255) / 256), 256, 0, stream>>>(src, dst, n4);
    };
    auto cvtT = [&](const float* Bm, _Float16* BTd, int K_, int N_) {
        dim3 g(N_ / 32, K_ / 32);
        k_cvtT_f16<<<g, 256, 0, stream>>>(Bm, BTd, K_, N_);
    };
    auto gemm = [&](const _Float16* Ah, const _Float16* BTd, const float* bias,
                    float* Cf, int M_, int K_, int N_, int relu) {
        dim3 g(N_ / 64, M_ / 64);
        k_gemm_f16w<<<g, 128, 0, stream>>>(Ah, BTd, bias, Cf, M_, K_, N_, relu);
    };

    cvt(A0f, A0h, Nd);
    cvtT(W1, BT, d, N);
    gemm(A0h, BT, nullptr, Xf, N, d, N, 0);   // X1 = A0 @ W1                [N x N]
    cvt(Lf, Lh, NN);
    cvtT(Xf, BT, N, N);
    gemm(Lh, BT, b1, Hf, N, N, N, 1);         // H1 = relu(L @ X1 + b1)
    cvt(Hf, Hh, NN);
    cvtT(W2, BT, N, N);
    gemm(Hh, BT, nullptr, Xf, N, N, N, 0);    // X2 = H1 @ W2
    cvtT(Xf, BT, N, N);
    gemm(Lh, BT, b2, Hf, N, N, N, 0);         // H2 = L @ X2 + b2
    cvt(Hf, Hh, NN);
    cvtT(W3, BT, N, d);
    gemm(Hh, BT, nullptr, Xf, N, N, d, 0);    // X3 = H2 @ W3                [N x d]
    cvtT(Xf, BT, N, d);
    gemm(Lh, BT, b3, LOG, N, N, d, 0);        // logits = L @ X3 + b3        [N x d]

    // gather + swap + copy feat -> out
    {
        long long tot4 = (long long)T * 7 * d / 4;
        k_assemble<<<(unsigned)((tot4 + 255) / 256), 256, 0, stream>>>(feat, nid, LOG, out, T, d);
    }
}
